// BiLSTM_CRF_83270825935413
// MI455X (gfx1250) — compile-verified, matrix operand order checked by reading, not run on hardware
//
#include <hip/hip_runtime.h>
#include <hip/hip_bf16.h>

// Problem constants (from reference): V=50000, E=256, HID=512, B=32, L=512, H=256, T=12
#define Bn   32
#define Ln   512
#define En   256
#define Hn   256
#define G4   1024        // 4*H
#define HIDn 512
#define Tn   12
#define TPAD 16
#define NEGV (-10000.0f)
#define START_TAG 10
#define STOP_TAG  11

typedef _Float16 half8   __attribute__((ext_vector_type(8)));
typedef _Float16 half16  __attribute__((ext_vector_type(16)));
typedef float    floatx8 __attribute__((ext_vector_type(8)));

union V16 { half16 v; half8 h[2]; };

__device__ __forceinline__ float sigmoidf(float x) { return 1.0f / (1.0f + __expf(-x)); }

// ---------------------------------------------------------------------------
// 1) Embedding gather -> f16   X[row(=b*L+l)][E]
// ---------------------------------------------------------------------------
__global__ void embed_gather(const int* __restrict__ sent,
                             const float* __restrict__ emb,
                             _Float16* __restrict__ X) {
    int row = blockIdx.x;          // 0..B*L-1
    int e   = threadIdx.x;         // 0..255
    int v   = sent[row];
    X[row * En + e] = (_Float16)emb[v * En + e];
}

// ---------------------------------------------------------------------------
// 2) f32 -> f16 weight conversion (generic)
// ---------------------------------------------------------------------------
__global__ void cvt_f16(const float* __restrict__ src, _Float16* __restrict__ dst, int n) {
    int i = blockIdx.x * blockDim.x + threadIdx.x;
    if (i < n) dst[i] = (_Float16)src[i];
}

// W_out [12][512] -> padded f16 [16][512] (rows 12..15 zero)
__global__ void wout_pad(const float* __restrict__ W, _Float16* __restrict__ dst) {
    int i = blockIdx.x * blockDim.x + threadIdx.x;   // 0..8191
    int t = i >> 9, h = i & 511;
    dst[i] = (t < Tn) ? (_Float16)W[t * HIDn + h] : (_Float16)0.0f;
}

// ---------------------------------------------------------------------------
// 3) Input GEMM: out[m][n] = sum_k X[m][k]*W[n][k] + bias[n]   (f16 in, f16 out)
//    M = B*L = 16384, N = 4H = 1024, K = E = 256.
//    Register-blocked: one 16x64 output strip (4 accumulators) per wave.
// ---------------------------------------------------------------------------
__global__ __launch_bounds__(256)
void input_gemm(const _Float16* __restrict__ X, const _Float16* __restrict__ W,
                const float* __restrict__ bias, _Float16* __restrict__ out) {
    int wave = (blockIdx.x * blockDim.x + threadIdx.x) >> 5;   // 16384 waves
    int lane = threadIdx.x & 31;
    int tm = wave >> 4;            // 1024 M-tiles
    int tg = wave & 15;            // 16 N-groups of 64 columns
    int m0 = tm * 16, n0 = tg * 64;
    int kh = lane >> 4, l16 = lane & 15;

    const _Float16* arow  = X + (m0 + l16) * En;
    const _Float16* brow0 = W + (n0 + l16) * En;

    floatx8 acc[4];
#pragma unroll
    for (int q = 0; q < 4; ++q) {
        float bv = bias[n0 + q * 16 + l16];
#pragma unroll
        for (int i = 0; i < 8; ++i) acc[q][i] = bv;
    }

#pragma unroll
    for (int k0 = 0; k0 < En; k0 += 32) {
        V16 va;
        va.h[0] = *(const half8*)(arow + k0 + kh * 8);
        va.h[1] = *(const half8*)(arow + k0 + 16 + kh * 8);
#pragma unroll
        for (int q = 0; q < 4; ++q) {
            const _Float16* brow = brow0 + q * 16 * En;
            V16 vb;
            vb.h[0] = *(const half8*)(brow + k0 + kh * 8);
            vb.h[1] = *(const half8*)(brow + k0 + 16 + kh * 8);
            acc[q] = __builtin_amdgcn_wmma_f32_16x16x32_f16(false, va.v, false, vb.v,
                                                            (short)0, acc[q], false, false);
        }
    }
#pragma unroll
    for (int q = 0; q < 4; ++q)
#pragma unroll
        for (int r = 0; r < 8; ++r)
            out[(m0 + r + 8 * kh) * G4 + n0 + q * 16 + l16] = (_Float16)acc[q][r];
}

// ---------------------------------------------------------------------------
// 4) Persistent LSTM recurrence, one direction, 16 batch rows per block.
//    grid = 2 blocks, blockDim = 512 (16 waves).
//    Per step t: g = xW[:,t,:] + h_prev @ W_hh^T ; gates -> c,h ; h kept in LDS.
//    K-loop hoisted: one LDS A-fragment load feeds 4 WMMA chains.
// ---------------------------------------------------------------------------
#define SHH_STRIDE 264    // halves, multiple of 8 -> 16B-aligned rows
#define SHG_STRIDE 1032   // halves

__global__ __launch_bounds__(512)
void lstm_recur(const _Float16* __restrict__ xW,    // [B][L][4H] f16 (bias included)
                const _Float16* __restrict__ Whh,   // [4H][H]    f16
                _Float16* __restrict__ hcat,        // [B][L][HID] f16
                int reverse, int dirOff) {
    __shared__ __align__(16) _Float16 sh_h[16 * SHH_STRIDE];  // h_prev (f16)
    __shared__ __align__(16) _Float16 sh_g[16 * SHG_STRIDE];  // gate preactivations

    int tid  = threadIdx.x;
    int lane = tid & 31;
    int w    = tid >> 5;           // wave 0..15
    int b0   = blockIdx.x * 16;    // batch base
    int kh   = lane >> 4, l16 = lane & 15;

    float creg[8];
#pragma unroll
    for (int p = 0; p < 8; ++p) creg[p] = 0.0f;
    for (int i = tid; i < 16 * SHH_STRIDE; i += 512) sh_h[i] = (_Float16)0.0f;
    __syncthreads();

    for (int s = 0; s < Ln; ++s) {
        int t = reverse ? (Ln - 1 - s) : s;

        // ---- recurrent GEMM: 64 N-tiles over 16 waves (4 chains per wave) ----
        floatx8 acc[4];
#pragma unroll
        for (int q = 0; q < 4; ++q) {
            int n0 = (w + q * 16) * 16;
#pragma unroll
            for (int r = 0; r < 8; ++r)
                acc[q][r] = (float)xW[((b0 + r + 8 * kh) * Ln + t) * G4 + n0 + l16];
        }

        const _Float16* arow = sh_h + l16 * SHH_STRIDE;
#pragma unroll
        for (int k0 = 0; k0 < Hn; k0 += 32) {
            V16 va;
            va.h[0] = *(const half8*)(arow + k0 + kh * 8);
            va.h[1] = *(const half8*)(arow + k0 + 16 + kh * 8);
#pragma unroll
            for (int q = 0; q < 4; ++q) {
                int n0 = (w + q * 16) * 16;
                const _Float16* brow = Whh + (n0 + l16) * Hn;
                V16 vb;
                vb.h[0] = *(const half8*)(brow + k0 + kh * 8);
                vb.h[1] = *(const half8*)(brow + k0 + 16 + kh * 8);
                acc[q] = __builtin_amdgcn_wmma_f32_16x16x32_f16(false, va.v, false, vb.v,
                                                                (short)0, acc[q], false, false);
            }
        }

        // prefetch next step's xW cachelines (global_prefetch_b8) while gates run
        if (s + 1 < Ln) {
            int t2 = reverse ? (Ln - 2 - s) : (s + 1);
#pragma unroll
            for (int q = 0; q < 4; ++q) {
                int n0 = (w + q * 16) * 16;
                __builtin_prefetch(&xW[((b0 + 8 * kh) * Ln + t2) * G4 + n0 + l16], 0, 1);
            }
        }

#pragma unroll
        for (int q = 0; q < 4; ++q) {
            int n0 = (w + q * 16) * 16;
#pragma unroll
            for (int r = 0; r < 8; ++r)
                sh_g[(r + 8 * kh) * SHG_STRIDE + n0 + l16] = (_Float16)acc[q][r];
        }
        __syncthreads();

        // ---- fused gates: c = sig(f)*c + sig(i)*tanh(g); h = sig(o)*tanh(c) ----
#pragma unroll
        for (int p = 0; p < 8; ++p) {
            int e  = tid + p * 512;     // 0..4095
            int bb = e >> 8;            // local batch row 0..15
            int j  = e & 255;           // hidden unit
            const _Float16* gr = sh_g + bb * SHG_STRIDE;
            float gi = (float)gr[j];
            float gf = (float)gr[256 + j];
            float gg = (float)gr[512 + j];
            float go = (float)gr[768 + j];
            float cc = sigmoidf(gf) * creg[p] + sigmoidf(gi) * tanhf(gg);
            creg[p]  = cc;
            float hh = sigmoidf(go) * tanhf(cc);
            sh_h[bb * SHH_STRIDE + j] = (_Float16)hh;
            hcat[((b0 + bb) * Ln + t) * HIDn + dirOff + j] = (_Float16)hh;
        }
        __syncthreads();
    }
}

// ---------------------------------------------------------------------------
// 5) Output projection: feats[m][n] = sum_k hcat[m][k]*Wout[n][k] + b_out[n]
//    M = 16384, N = 16 (padded from 12), K = 512.  One M-tile per wave.
// ---------------------------------------------------------------------------
__global__ __launch_bounds__(256)
void proj_gemm(const _Float16* __restrict__ H, const _Float16* __restrict__ Wo,
               const float* __restrict__ bo, float* __restrict__ feats) {
    int wave = (blockIdx.x * blockDim.x + threadIdx.x) >> 5;   // 0..1023
    int lane = threadIdx.x & 31;
    int m0 = wave * 16;
    int kh = lane >> 4, l16 = lane & 15;

    const _Float16* arow = H + (m0 + l16) * HIDn;
    const _Float16* brow = Wo + l16 * HIDn;

    float bv = (l16 < Tn) ? bo[l16] : 0.0f;
    floatx8 acc;
#pragma unroll
    for (int i = 0; i < 8; ++i) acc[i] = bv;

#pragma unroll
    for (int k0 = 0; k0 < HIDn; k0 += 32) {
        V16 va, vb;
        va.h[0] = *(const half8*)(arow + k0 + kh * 8);
        va.h[1] = *(const half8*)(arow + k0 + 16 + kh * 8);
        vb.h[0] = *(const half8*)(brow + k0 + kh * 8);
        vb.h[1] = *(const half8*)(brow + k0 + 16 + kh * 8);
        acc = __builtin_amdgcn_wmma_f32_16x16x32_f16(false, va.v, false, vb.v,
                                                     (short)0, acc, false, false);
    }
#pragma unroll
    for (int r = 0; r < 8; ++r)
        feats[(m0 + r + 8 * kh) * TPAD + l16] = acc[r];
}

// ---------------------------------------------------------------------------
// 6) Viterbi decode: 1 wave, lane = batch. fv in registers, trans in LDS.
// ---------------------------------------------------------------------------
__global__ void viterbi(const float* __restrict__ feats,     // [B*L][TPAD]
                        const float* __restrict__ trans,     // [T][T] (next,prev)
                        int* __restrict__ bp,                // [L][B][T]
                        float* __restrict__ out) {           // score[B] ++ path[B][L]
    __shared__ float sT[Tn * Tn];
    int lane = threadIdx.x;
    for (int i = lane; i < Tn * Tn; i += 32) sT[i] = trans[i];
    __syncthreads();

    int b = lane;   // 0..31
    float fv[Tn];
#pragma unroll
    for (int n = 0; n < Tn; ++n) fv[n] = NEGV;
    fv[START_TAG] = 0.0f;

    for (int t = 0; t < Ln; ++t) {
        float nfv[Tn];
#pragma unroll
        for (int nx = 0; nx < Tn; ++nx) {
            float best = -3.4e38f; int arg = 0;
#pragma unroll
            for (int pv = 0; pv < Tn; ++pv) {
                float sc = fv[pv] + sT[nx * Tn + pv];
                if (sc > best) { best = sc; arg = pv; }
            }
            bp[(t * Bn + b) * Tn + nx] = arg;
            nfv[nx] = best + feats[(b * Ln + t) * TPAD + nx];
        }
#pragma unroll
        for (int nx = 0; nx < Tn; ++nx) fv[nx] = nfv[nx];
    }

    float best = -3.4e38f; int bt = 0;
#pragma unroll
    for (int nx = 0; nx < Tn; ++nx) {
        float sc = fv[nx] + sT[STOP_TAG * Tn + nx];
        if (sc > best) { best = sc; bt = nx; }
    }
    out[b] = best;

    int tag = bt;
    for (int t = Ln - 1; t >= 0; --t) {
        out[Bn + b * Ln + t] = (float)tag;
        tag = bp[(t * Bn + b) * Tn + tag];
    }
}

// ---------------------------------------------------------------------------
// Host launcher
// ---------------------------------------------------------------------------
static inline size_t alignup(size_t x) { return (x + 255) & ~(size_t)255; }

extern "C" void kernel_launch(void* const* d_in, const int* in_sizes, int n_in,
                              void* d_out, int out_size, void* d_ws, size_t ws_size,
                              hipStream_t stream) {
    (void)in_sizes; (void)n_in; (void)out_size; (void)ws_size;
    const int*   sentence = (const int*)  d_in[0];
    const float* emb      = (const float*)d_in[1];
    const float* W_ih_f   = (const float*)d_in[2];
    const float* W_hh_f   = (const float*)d_in[3];
    const float* b_f      = (const float*)d_in[4];
    const float* W_ih_b   = (const float*)d_in[5];
    const float* W_hh_b   = (const float*)d_in[6];
    const float* b_b      = (const float*)d_in[7];
    const float* W_out    = (const float*)d_in[8];
    const float* b_out    = (const float*)d_in[9];
    const float* trans    = (const float*)d_in[10];
    float* out = (float*)d_out;

    // Workspace layout
    char* ws = (char*)d_ws;
    size_t o = 0;
    const size_t M = (size_t)Bn * Ln;                             // 16384
    _Float16* x16     = (_Float16*)(ws + o); o += alignup(M * En * 2);
    _Float16* wih_f16 = (_Float16*)(ws + o); o += alignup((size_t)G4 * En * 2);
    _Float16* wih_b16 = (_Float16*)(ws + o); o += alignup((size_t)G4 * En * 2);
    _Float16* whh_f16 = (_Float16*)(ws + o); o += alignup((size_t)G4 * Hn * 2);
    _Float16* whh_b16 = (_Float16*)(ws + o); o += alignup((size_t)G4 * Hn * 2);
    _Float16* wout16  = (_Float16*)(ws + o); o += alignup((size_t)TPAD * HIDn * 2);
    _Float16* xw_f    = (_Float16*)(ws + o); o += alignup(M * G4 * 2);
    _Float16* xw_b    = (_Float16*)(ws + o); o += alignup(M * G4 * 2);
    _Float16* hcat    = (_Float16*)(ws + o); o += alignup(M * HIDn * 2);
    float*    feats   = (float*)   (ws + o); o += alignup(M * TPAD * 4);
    int*      bp      = (int*)     (ws + o); o += alignup((size_t)Ln * Bn * Tn * 4);

    // 1) embedding gather -> f16
    embed_gather<<<(int)M, En, 0, stream>>>(sentence, emb, x16);

    // 2) weight conversions
    const int NW = G4 * En;   // 262144
    cvt_f16<<<(NW + 255) / 256, 256, 0, stream>>>(W_ih_f, wih_f16, NW);
    cvt_f16<<<(NW + 255) / 256, 256, 0, stream>>>(W_ih_b, wih_b16, NW);
    cvt_f16<<<(NW + 255) / 256, 256, 0, stream>>>(W_hh_f, whh_f16, NW);
    cvt_f16<<<(NW + 255) / 256, 256, 0, stream>>>(W_hh_b, whh_b16, NW);
    wout_pad<<<(TPAD * HIDn + 255) / 256, 256, 0, stream>>>(W_out, wout16);

    // 3) input GEMMs (16384x1024x256), 16x64 strip per wave: 16384 waves
    input_gemm<<<2048, 256, 0, stream>>>(x16, wih_f16, b_f, xw_f);
    input_gemm<<<2048, 256, 0, stream>>>(x16, wih_b16, b_b, xw_b);

    // 4) recurrences (persistent; 2 blocks x 16 batch rows each)
    lstm_recur<<<2, 512, 0, stream>>>(xw_f, whh_f16, hcat, /*reverse=*/0, /*dirOff=*/0);
    lstm_recur<<<2, 512, 0, stream>>>(xw_b, whh_b16, hcat, /*reverse=*/1, /*dirOff=*/Hn);

    // 5) output projection (16384x16x512), one M-tile per wave: 1024 waves
    proj_gemm<<<128, 256, 0, stream>>>(hcat, wout16, b_out, feats);

    // 6) Viterbi decode + backtrack
    viterbi<<<1, 32, 0, stream>>>(feats, trans, bp, out);
}
